// CantorLinear_91104846283071
// MI455X (gfx1250) — matrix operand order repaired and checked by reading, not compiled, not run on hardware
//
#include <hip/hip_runtime.h>
#include <hip/hip_bf16.h>

typedef __bf16 bf16;
typedef __attribute__((ext_vector_type(16))) __bf16 v16bf;
typedef __attribute__((ext_vector_type(8)))  __bf16 v8bf;
typedef __attribute__((ext_vector_type(4)))  __bf16 v4bf;
typedef __attribute__((ext_vector_type(8)))  float  v8f;

#define M_TOTAL 16384
#define N_TOTAL 2048
#define K_TOTAL 2048
#define BM 128
#define BN 128
#define BK 64
#define KTILES (K_TOTAL / BK)
// LDS row stride in elements: 64 + 8 -> 144 bytes -> bank stride 36 (mod 64),
// gcd(36,64)=4, so 16 lanes hit 16 distinct 4-bank spans: conflict-free b128.
#define LDSS (BK + 8)
#define STAGE_BYTES (BM * LDSS * 2)   // 18432 B per buffer

// ---------------------------------------------------------------------------
// Prologue: W_eff = (weight * mask) downconverted to bf16, stored in d_ws.
// ---------------------------------------------------------------------------
__global__ __launch_bounds__(256)
void prep_weff_kernel(const float* __restrict__ w, const float* __restrict__ m,
                      bf16* __restrict__ weff) {
  const int i = (blockIdx.x * 256 + threadIdx.x) * 4;
  const float4 wv = *(const float4*)(w + i);
  const float4 mv = *(const float4*)(m + i);
  v4bf o;
  o.x = (bf16)(wv.x * mv.x);
  o.y = (bf16)(wv.y * mv.y);
  o.z = (bf16)(wv.z * mv.z);
  o.w = (bf16)(wv.w * mv.w);
  *(v4bf*)(weff + i) = o;
}

// ---------------------------------------------------------------------------
// Main GEMM: out[M][N] = x[M][K] * W_eff[N][K]^T + bias[N]
// 128x128 block tile, BK=64, 8 wave32s (4 in M, 2 in N), WMMA bf16 -> f32.
// A tile: global f32 -> VGPR -> cvt bf16 -> ds_store (needs conversion).
// B tile: GLOBAL_LOAD_ASYNC_TO_LDS_B128 (already bf16, bypasses VGPRs,
//         tracked with ASYNCcnt, fenced by s_wait_asynccnt + barrier).
// ---------------------------------------------------------------------------
__global__ __launch_bounds__(256)
void cantor_gemm_kernel(const float* __restrict__ x,
                        const bf16*  __restrict__ weff,
                        const float* __restrict__ bias,
                        float* __restrict__ out) {
  __shared__ bf16 As[2][BM * LDSS];
  __shared__ bf16 Bs[2][BM * LDSS];

  const int tid   = threadIdx.x;
  const int lane  = tid & 31;
  const int wave  = tid >> 5;
  const int waveM = wave & 3;   // 0..3 -> 32-row slab
  const int waveN = wave >> 2;  // 0..1 -> 64-col slab

  const int mBlock = blockIdx.y * BM;
  const int nBlock = blockIdx.x * BN;

  // Cooperative global->LDS mapping: 8 threads per row (16B each), 32 rows/pass.
  const int lrow = tid >> 3;        // 0..31
  const int lcol = (tid & 7) * 8;   // 0,8,...,56 (elements)

  const float* xg = x    + (size_t)(mBlock + lrow) * K_TOTAL + lcol;
  const bf16*  wg = weff + (size_t)(nBlock + lrow) * K_TOTAL + lcol;

  // LDS byte offset of this thread's B-tile slot (flat->LDS truncates to
  // addr[31:0] == wave-relative LDS offset, per ISA 10.2 aperture mapping).
  const unsigned bsOff =
      (unsigned)(uintptr_t)(&Bs[0][0]) + (unsigned)((lrow * LDSS + lcol) * 2);

  float4 aReg[4][2];

  auto load_globalA = [&](int t) {
    const float* xp0 = xg + (size_t)t * BK;
#pragma unroll
    for (int p = 0; p < 4; ++p) {
      const float* xp = xp0 + (size_t)p * 32 * K_TOTAL;
      aReg[p][0] = *(const float4*)(xp);
      aReg[p][1] = *(const float4*)(xp + 4);
    }
  };

  // Async DMA of B tile (bf16 weights) straight into LDS buffer `buf`.
  auto issue_asyncB = [&](int t, int buf) {
    const bf16* wp0 = wg + (size_t)t * BK;
    const unsigned l0 = bsOff + (unsigned)buf * STAGE_BYTES;
#pragma unroll
    for (int p = 0; p < 4; ++p) {
      const unsigned long long ga =
          (unsigned long long)(uintptr_t)(wp0 + (size_t)p * 32 * K_TOTAL);
      const unsigned lo = l0 + (unsigned)(p * 32 * LDSS * 2);
      asm volatile("global_load_async_to_lds_b128 %0, %1, off"
                   :: "v"(lo), "v"(ga)
                   : "memory");
    }
  };

  auto store_stageA = [&](bf16* Adst) {
#pragma unroll
    for (int p = 0; p < 4; ++p) {
      const int row = lrow + p * 32;
      v8bf av;
      av[0] = (bf16)aReg[p][0].x; av[1] = (bf16)aReg[p][0].y;
      av[2] = (bf16)aReg[p][0].z; av[3] = (bf16)aReg[p][0].w;
      av[4] = (bf16)aReg[p][1].x; av[5] = (bf16)aReg[p][1].y;
      av[6] = (bf16)aReg[p][1].z; av[7] = (bf16)aReg[p][1].w;
      *(v8bf*)(Adst + row * LDSS + lcol) = av;
    }
  };

  // Prime pipeline: tile 0 into LDS buffer 0.
  issue_asyncB(0, 0);
  load_globalA(0);
  store_stageA(As[0]);
  asm volatile("s_wait_asynccnt 0x0" ::: "memory");
  __syncthreads();

  v8f acc[2][4] = {};

  const int r15   = lane & 15;        // N (B/C/D) or M (A) index within tile
  const int hselA = (lane >> 4) * 8;  // A: K phase 0 or 8
  const int hselB = (lane >> 4) * 8;  // B: K phase 0 or 8

  int cur = 0;
  for (int t = 0; t < KTILES; ++t) {
    if (t + 1 < KTILES) {
      issue_asyncB(t + 1, cur ^ 1);   // DMA next B tile into other buffer
      load_globalA(t + 1);            // prefetch next A tile to VGPRs
    }

    const bf16* Acur = As[cur];
    const bf16* Bcur = Bs[cur];
#pragma unroll
    for (int ks = 0; ks < 2; ++ks) {          // two K=32 steps per BK=64
      const int kk = ks * 32;
      v16bf afrag[2], bfrag[4];
#pragma unroll
      for (int i = 0; i < 2; ++i) {
        // A 16x32 bf16 layout: lanes 0-15 K{0..7,16..23}, lanes 16-31 K{8..15,24..31}
        const bf16* p = Acur + (waveM * 32 + i * 16 + r15) * LDSS + kk + hselA;
        union { v16bf v; v8bf h[2]; } u;
        u.h[0] = *(const v8bf*)(p);
        u.h[1] = *(const v8bf*)(p + 16);
        afrag[i] = u.v;
      }
#pragma unroll
      for (int j = 0; j < 4; ++j) {
        // B 32x16 bf16 layout: lane n holds col N=n, K{0..7,16..23}; lane n+16: K{8..15,24..31}
        const bf16* p = Bcur + (waveN * 64 + j * 16 + r15) * LDSS + kk + hselB;
        union { v16bf v; v8bf h[2]; } u;
        u.h[0] = *(const v8bf*)(p);
        u.h[1] = *(const v8bf*)(p + 16);
        bfrag[j] = u.v;
      }
#pragma unroll
      for (int i = 0; i < 2; ++i)
#pragma unroll
        for (int j = 0; j < 4; ++j)
          acc[i][j] = __builtin_amdgcn_wmma_f32_16x16x32_bf16(
              false, afrag[i], false, bfrag[j],
              (short)0, acc[i][j], false, false);
    }

    if (t + 1 < KTILES) {
      const int nxt = cur ^ 1;
      store_stageA(As[nxt]);                       // other buffer: safe pre-barrier
      asm volatile("s_wait_asynccnt 0x0" ::: "memory");  // B DMA landed in LDS
      __syncthreads();
      cur = nxt;
    }
  }

  // Epilogue: D layout = VGPR r -> row r (lanes 0-15) / row 8+r (lanes 16-31),
  // N = lane & 15. Add bias (column-only), store f32.
  const int hiRow = (lane >> 4) * 8;
#pragma unroll
  for (int i = 0; i < 2; ++i) {
    const int mBase = mBlock + waveM * 32 + i * 16 + hiRow;
#pragma unroll
    for (int j = 0; j < 4; ++j) {
      const int nCol = nBlock + waveN * 64 + j * 16 + r15;
      const float bv = bias[nCol];
#pragma unroll
      for (int r = 0; r < 8; ++r) {
        out[(size_t)(mBase + r) * N_TOTAL + nCol] = acc[i][j][r] + bv;
      }
    }
  }
}

// ---------------------------------------------------------------------------
// Launcher. Inputs (setup_inputs order): x[4*4096*2048] f32, weight[2048*2048]
// f32, bias[2048] f32, mask[2048*2048] f32. Output: f32 [16384*2048].
// d_ws holds W_eff as bf16 (8 MB), recomputed every call (deterministic).
// ---------------------------------------------------------------------------
extern "C" void kernel_launch(void* const* d_in, const int* in_sizes, int n_in,
                              void* d_out, int out_size, void* d_ws, size_t ws_size,
                              hipStream_t stream) {
  const float* x    = (const float*)d_in[0];
  const float* w    = (const float*)d_in[1];
  const float* bias = (const float*)d_in[2];
  const float* mask = (const float*)d_in[3];
  float* out = (float*)d_out;
  bf16* weff = (bf16*)d_ws;

  prep_weff_kernel<<<(N_TOTAL * K_TOTAL) / (256 * 4), 256, 0, stream>>>(w, mask, weff);

  dim3 grid(N_TOTAL / BN, M_TOTAL / BM);
  cantor_gemm_kernel<<<grid, 256, 0, stream>>>(x, weff, bias, out);
}
